// ASDFEncoder_67740224193368
// MI455X (gfx1250) — compile-verified, matrix operand order checked by reading, not run on hardware
//
#include <hip/hip_runtime.h>
#include <hip/hip_bf16.h>

// ---------------- problem constants (from reference) ----------------
#define BATCH 16
#define NPTS  16384
#define NC    40          // centers per batch
#define KNN   410         // ceil(N/C)
#define BCC   (BATCH*NC)  // 640 total centers (multiple of 16 -> no M guards)
#define HID   192
#define HEADS 6
#define HD    32          // HID/HEADS
#define DEPTH 6
#define MSGD  51          // 3 + 48 sinusoidal
#define QKVD  576
#define FFD   768

typedef __bf16 v16bf __attribute__((ext_vector_type(16)));
typedef float  v8f   __attribute__((ext_vector_type(8)));

union ABfrag { v16bf v; unsigned u[8]; };
union AccFrag { v8f v; float f[8]; };

// ---------------- helpers ----------------
__device__ __forceinline__ unsigned rne_bf16_bits(float f) {
  unsigned u = __float_as_uint(f);
  return (u + 0x7FFFu + ((u >> 16) & 1u)) >> 16;   // round-to-nearest-even bf16
}
__device__ __forceinline__ unsigned pack2bf(float lo, float hi) {
  return rne_bf16_bits(lo) | (rne_bf16_bits(hi) << 16);
}
// order-preserving float->uint key (atomicMax-based max aggregation)
__device__ __forceinline__ unsigned fkey(float f) {
  unsigned u = __float_as_uint(f);
  return (u & 0x80000000u) ? ~u : (u | 0x80000000u);
}
__device__ __forceinline__ float fkey_dec(unsigned k) {
  unsigned u = (k & 0x80000000u) ? (k & 0x7FFFFFFFu) : ~k;
  return __uint_as_float(u);
}

__device__ __forceinline__ int block_count_reduce(int v, int* sbuf) {
  int tid = threadIdx.x;
  sbuf[tid] = v; __syncthreads();
  for (int s = 128; s > 0; s >>= 1) { if (tid < s) sbuf[tid] += sbuf[tid + s]; __syncthreads(); }
  int r = sbuf[0]; __syncthreads();
  return r;
}

#define WMMA_BF16(A, B, C) \
  __builtin_amdgcn_wmma_f32_16x16x32_bf16(false, (A), false, (B), (short)0, (C), false, false)

// ---------------- weight pre-pack: W[KxN] f32 -> Wp[n][k/2] bf16-pair dwords ----------------
__global__ void pack_weight_kernel(const float* __restrict__ W, unsigned* __restrict__ Wp,
                                   int K, int N, int Kpu, int total)
{
  int i = blockIdx.x * blockDim.x + threadIdx.x;
  if (i >= total) return;
  int n = i / Kpu, kp = i - n * Kpu;
  int k0 = 2 * kp, k1 = 2 * kp + 1;
  float w0 = (k0 < K) ? W[(size_t)k0 * N + n] : 0.f;
  float w1 = (k1 < K) ? W[(size_t)k1 * N + n] : 0.f;
  Wp[i] = pack2bf(w0, w1);
}

// ---------------- KNN: radix-select 410 nearest per center ----------------
__global__ __launch_bounds__(256)
void knn_kernel(const float* __restrict__ pc, const int* __restrict__ idx,
                int* __restrict__ nbr, float* __restrict__ centers)
{
  extern __shared__ unsigned ubuf[];        // 16384 uint d2 keys (64 KB dynamic LDS)
  __shared__ int sred[256];
  __shared__ float cpos[3];
  __shared__ int scnt[2];
  int bc = blockIdx.x;
  int b = bc / NC, c = bc % NC;
  int tid = threadIdx.x;
  if (tid == 0) {
    int j = idx[b * NC + c];
    const float* q = pc + ((size_t)b * NPTS + j) * 3;
    cpos[0] = q[0]; cpos[1] = q[1]; cpos[2] = q[2];
    centers[bc * 3 + 0] = q[0]; centers[bc * 3 + 1] = q[1]; centers[bc * 3 + 2] = q[2];
    scnt[0] = 0; scnt[1] = 0;
  }
  __syncthreads();
  float cx = cpos[0], cy = cpos[1], cz = cpos[2];
  for (int i = tid; i < NPTS; i += 256) {
    const float* q = pc + ((size_t)b * NPTS + i) * 3;
    float dx = q[0] - cx, dy = q[1] - cy, dz = q[2] - cz;
    ubuf[i] = __float_as_uint(dx * dx + dy * dy + dz * dz);  // >=0 -> monotone as uint
  }
  __syncthreads();
  unsigned lo = 0u, hi = 0x7F800000u;      // V = 410th-smallest key
  while (lo < hi) {
    unsigned mid = lo + ((hi - lo) >> 1);
    int cnt = 0;
    for (int i = tid; i < NPTS; i += 256) cnt += (ubuf[i] <= mid) ? 1 : 0;
    int tot = block_count_reduce(cnt, sred);
    if (tot >= KNN) hi = mid; else lo = mid + 1;
  }
  unsigned V = lo;
  for (int i = tid; i < NPTS; i += 256)
    if (ubuf[i] < V) { int p = atomicAdd(&scnt[0], 1); nbr[(size_t)bc * KNN + p] = i; }
  __syncthreads();
  int base = scnt[0];
  for (int i = tid; i < NPTS; i += 256)
    if (ubuf[i] == V) {
      int p = atomicAdd(&scnt[1], 1) + base;
      if (p < KNN) nbr[(size_t)bc * KNN + p] = i;
    }
}

// ---------------- fused PointConv local MLP (4 WMMA layers + max) ----------------
// 16x32 strip per wave, 4 independent accumulator chains (dual-N x K-parity split)
__device__ __forceinline__
void pc_layer(const unsigned short* bin, unsigned short* bout,
              const unsigned* __restrict__ Wp, int Kpu, const float* __restrict__ Bv,
              int Kp /*padded K, mult of 32*/, int Nd, int lane, bool relu)
{
  int half = lane >> 4, l15 = lane & 15;
  const unsigned short* arow = bin + l15 * HID + 8 * half;
  for (int nt = 0; nt < Nd / 32; ++nt) {
    int n0 = nt * 32 + l15, n1 = n0 + 16;
    const unsigned* wrow0 = Wp + (size_t)n0 * Kpu + 4 * half;
    const unsigned* wrow1 = Wp + (size_t)n1 * Kpu + 4 * half;
    AccFrag a0x, a0y, a1x, a1y;
#pragma unroll
    for (int j = 0; j < 8; ++j) { a0x.f[j] = 0.f; a0y.f[j] = 0.f; a1x.f[j] = 0.f; a1y.f[j] = 0.f; }
    int kt = 0;
    for (; kt + 64 <= Kp; kt += 64) {
      ABfrag aA, aB, w0A, w0B, w1A, w1B;
      *(uint4*)&aA.u[0]  = *(const uint4*)(arow + kt);
      *(uint4*)&aA.u[4]  = *(const uint4*)(arow + kt + 16);
      *(uint4*)&aB.u[0]  = *(const uint4*)(arow + kt + 32);
      *(uint4*)&aB.u[4]  = *(const uint4*)(arow + kt + 48);
      *(uint4*)&w0A.u[0] = *(const uint4*)(wrow0 + (kt >> 1));
      *(uint4*)&w0A.u[4] = *(const uint4*)(wrow0 + (kt >> 1) + 8);
      *(uint4*)&w0B.u[0] = *(const uint4*)(wrow0 + (kt >> 1) + 16);
      *(uint4*)&w0B.u[4] = *(const uint4*)(wrow0 + (kt >> 1) + 24);
      *(uint4*)&w1A.u[0] = *(const uint4*)(wrow1 + (kt >> 1));
      *(uint4*)&w1A.u[4] = *(const uint4*)(wrow1 + (kt >> 1) + 8);
      *(uint4*)&w1B.u[0] = *(const uint4*)(wrow1 + (kt >> 1) + 16);
      *(uint4*)&w1B.u[4] = *(const uint4*)(wrow1 + (kt >> 1) + 24);
      a0x.v = WMMA_BF16(aA.v, w0A.v, a0x.v);
      a1x.v = WMMA_BF16(aA.v, w1A.v, a1x.v);
      a0y.v = WMMA_BF16(aB.v, w0B.v, a0y.v);
      a1y.v = WMMA_BF16(aB.v, w1B.v, a1y.v);
    }
    if (kt < Kp) {                           // single remaining 32-chunk
      ABfrag aA, w0A, w1A;
      *(uint4*)&aA.u[0]  = *(const uint4*)(arow + kt);
      *(uint4*)&aA.u[4]  = *(const uint4*)(arow + kt + 16);
      *(uint4*)&w0A.u[0] = *(const uint4*)(wrow0 + (kt >> 1));
      *(uint4*)&w0A.u[4] = *(const uint4*)(wrow0 + (kt >> 1) + 8);
      *(uint4*)&w1A.u[0] = *(const uint4*)(wrow1 + (kt >> 1));
      *(uint4*)&w1A.u[4] = *(const uint4*)(wrow1 + (kt >> 1) + 8);
      a0x.v = WMMA_BF16(aA.v, w0A.v, a0x.v);
      a1x.v = WMMA_BF16(aA.v, w1A.v, a1x.v);
    }
    AccFrag acc0, acc1;
    acc0.v = a0x.v + a0y.v;
    acc1.v = a1x.v + a1y.v;
    float b0 = Bv[n0], b1 = Bv[n1];
#pragma unroll
    for (int j = 0; j < 8; ++j) {
      float v0 = acc0.f[j] + b0;
      float v1 = acc1.f[j] + b1;
      if (relu) { v0 = fmaxf(v0, 0.f); v1 = fmaxf(v1, 0.f); }
      bout[(j + 8 * half) * HID + n0] = (unsigned short)rne_bf16_bits(v0);
      bout[(j + 8 * half) * HID + n1] = (unsigned short)rne_bf16_bits(v1);
    }
  }
}

__device__ __forceinline__
void pc_layer_final(const unsigned short* bin,
                    const unsigned* __restrict__ Wp, int Kpu, const float* __restrict__ Bv,
                    int Kp, int Nd, int lane, unsigned* featk, int bc)
{
  int half = lane >> 4, l15 = lane & 15;
  const unsigned short* arow = bin + l15 * HID + 8 * half;
  for (int nt = 0; nt < Nd / 32; ++nt) {
    int n0 = nt * 32 + l15, n1 = n0 + 16;
    const unsigned* wrow0 = Wp + (size_t)n0 * Kpu + 4 * half;
    const unsigned* wrow1 = Wp + (size_t)n1 * Kpu + 4 * half;
    AccFrag a0x, a0y, a1x, a1y;
#pragma unroll
    for (int j = 0; j < 8; ++j) { a0x.f[j] = 0.f; a0y.f[j] = 0.f; a1x.f[j] = 0.f; a1y.f[j] = 0.f; }
    int kt = 0;
    for (; kt + 64 <= Kp; kt += 64) {
      ABfrag aA, aB, w0A, w0B, w1A, w1B;
      *(uint4*)&aA.u[0]  = *(const uint4*)(arow + kt);
      *(uint4*)&aA.u[4]  = *(const uint4*)(arow + kt + 16);
      *(uint4*)&aB.u[0]  = *(const uint4*)(arow + kt + 32);
      *(uint4*)&aB.u[4]  = *(const uint4*)(arow + kt + 48);
      *(uint4*)&w0A.u[0] = *(const uint4*)(wrow0 + (kt >> 1));
      *(uint4*)&w0A.u[4] = *(const uint4*)(wrow0 + (kt >> 1) + 8);
      *(uint4*)&w0B.u[0] = *(const uint4*)(wrow0 + (kt >> 1) + 16);
      *(uint4*)&w0B.u[4] = *(const uint4*)(wrow0 + (kt >> 1) + 24);
      *(uint4*)&w1A.u[0] = *(const uint4*)(wrow1 + (kt >> 1));
      *(uint4*)&w1A.u[4] = *(const uint4*)(wrow1 + (kt >> 1) + 8);
      *(uint4*)&w1B.u[0] = *(const uint4*)(wrow1 + (kt >> 1) + 16);
      *(uint4*)&w1B.u[4] = *(const uint4*)(wrow1 + (kt >> 1) + 24);
      a0x.v = WMMA_BF16(aA.v, w0A.v, a0x.v);
      a1x.v = WMMA_BF16(aA.v, w1A.v, a1x.v);
      a0y.v = WMMA_BF16(aB.v, w0B.v, a0y.v);
      a1y.v = WMMA_BF16(aB.v, w1B.v, a1y.v);
    }
    if (kt < Kp) {
      ABfrag aA, w0A, w1A;
      *(uint4*)&aA.u[0]  = *(const uint4*)(arow + kt);
      *(uint4*)&aA.u[4]  = *(const uint4*)(arow + kt + 16);
      *(uint4*)&w0A.u[0] = *(const uint4*)(wrow0 + (kt >> 1));
      *(uint4*)&w0A.u[4] = *(const uint4*)(wrow0 + (kt >> 1) + 8);
      *(uint4*)&w1A.u[0] = *(const uint4*)(wrow1 + (kt >> 1));
      *(uint4*)&w1A.u[4] = *(const uint4*)(wrow1 + (kt >> 1) + 8);
      a0x.v = WMMA_BF16(aA.v, w0A.v, a0x.v);
      a1x.v = WMMA_BF16(aA.v, w1A.v, a1x.v);
    }
    AccFrag acc0, acc1;
    acc0.v = a0x.v + a0y.v;
    acc1.v = a1x.v + a1y.v;
    float o0 = -1e30f, o1 = -1e30f;
    float b0 = Bv[n0], b1 = Bv[n1];
#pragma unroll
    for (int j = 0; j < 8; ++j) {
      o0 = fmaxf(o0, acc0.f[j] + b0);
      o1 = fmaxf(o1, acc1.f[j] + b1);
    }
    o0 = fmaxf(o0, __shfl_xor(o0, 16, 32));     // merge row halves (M 0..7 | 8..15)
    o1 = fmaxf(o1, __shfl_xor(o1, 16, 32));
    if (half == 0) {
      atomicMax(&featk[(size_t)bc * HID + n0], fkey(o0));
      atomicMax(&featk[(size_t)bc * HID + n1], fkey(o1));
    }
  }
}

__global__ __launch_bounds__(128)
void pointconv_kernel(const float* __restrict__ pc, const int* __restrict__ nbr,
                      const float* __restrict__ centers, const float* __restrict__ basis,
                      const unsigned* Wp1, const float* B1, const unsigned* Wp2, const float* B2,
                      const unsigned* Wp3, const float* B3, const unsigned* Wp4, const float* B4,
                      unsigned* __restrict__ featk, int totalTiles)
{
  __shared__ unsigned short lbuf[4][2][16 * HID];   // 48 KB: per-wave private ping-pong
  int wv = threadIdx.x >> 5;
  int lane = threadIdx.x & 31;
  int tile = blockIdx.x * 4 + wv;
  if (tile >= totalTiles) return;                   // wave-uniform: EXEC stays all-ones
  int bc = tile / 26, rt = tile - bc * 26;
  int b = bc / NC;
  int half = lane >> 4, l15 = lane & 15;
  unsigned short* buf0 = &lbuf[wv][0][0];
  unsigned short* buf1 = &lbuf[wv][1][0];

  // build 16x51 message tile (rel + sin/cos embed), zero-pad cols 51..63
  if (half == 0) {
    int r = rt * 16 + l15; if (r > KNN - 1) r = KNN - 1;  // clamp: dup rows don't hurt max
    int pj = nbr[(size_t)bc * KNN + r];
    const float* q = pc + ((size_t)b * NPTS + pj) * 3;
    float rx = q[0] - centers[bc * 3 + 0];
    float ry = q[1] - centers[bc * 3 + 1];
    float rz = q[2] - centers[bc * 3 + 2];
    unsigned short* row = buf0 + l15 * HID;
    row[0] = (unsigned short)rne_bf16_bits(rx);
    row[1] = (unsigned short)rne_bf16_bits(ry);
    row[2] = (unsigned short)rne_bf16_bits(rz);
#pragma unroll
    for (int e = 0; e < 24; ++e) {
      float p = rx * basis[e] + ry * basis[24 + e] + rz * basis[48 + e];
      row[3 + e]  = (unsigned short)rne_bf16_bits(sinf(p));
      row[27 + e] = (unsigned short)rne_bf16_bits(cosf(p));
    }
    for (int k = MSGD; k < 64; ++k) row[k] = 0;
  }
  // per-wave LDS ops are in-order (DScnt tracking): no barriers needed
  pc_layer(buf0, buf1, Wp1, 32, B1, 64, 96, lane, true);        // 51(pad 64) -> 96
  pc_layer(buf1, buf0, Wp2, 48, B2, 96, 96, lane, true);        // 96 -> 96
  pc_layer(buf0, buf1, Wp3, 48, B3, 96, HID, lane, true);       // 96 -> 192
  pc_layer_final(buf1, Wp4, 96, B4, HID, HID, lane, featk, bc); // 192 -> 192, max
}

// decode uint max-keys -> bf16 activation mirror
__global__ void decode_feat_kernel(const unsigned* __restrict__ keys,
                                   unsigned short* __restrict__ outb, int n)
{
  int i = blockIdx.x * blockDim.x + threadIdx.x;
  if (i < n) outb[i] = (unsigned short)rne_bf16_bits(fkey_dec(keys[i]));
}

// ---------------- generic WMMA GEMM (16x32 strip per wave, 4 accumulator chains) ----------------
// Ab: bf16 [M x ldab] (K zero-padded to mult of 32), Wp: packed bf16 pairs [N x Kpu]
// outf (opt): f32 [M x ldo], outb (opt): bf16 [M x ldob]. act: 0=none 1=relu 2=gelu
__global__ __launch_bounds__(256)
void gemm_wmma_kernel(const unsigned short* __restrict__ Ab, int ldab,
                      const unsigned* __restrict__ Wp, int Kpu,
                      const float* __restrict__ bias, const float* __restrict__ res,
                      float* __restrict__ outf, int ldo,
                      unsigned short* __restrict__ outb, int ldob,
                      int M, int K, int N, int act, int Nt2, int tiles)
{
  int wv = threadIdx.x >> 5;
  int tile = blockIdx.x * 8 + wv;
  if (tile >= tiles) return;                 // wave-uniform exit
  int lane = threadIdx.x & 31;
  int half = lane >> 4, l15 = lane & 15;
  int tm = tile / Nt2, tn = tile - tm * Nt2;
  int m = tm * 16 + l15;                     // M always mult of 16 here
  int n0 = tn * 32 + l15;
  int n1 = n0 + 16;
  int nn0 = (n0 < N) ? n0 : 0;               // clamp W rows for tail-N tiles
  int nn1 = (n1 < N) ? n1 : 0;
  const unsigned short* arow = Ab + (size_t)m * ldab + 8 * half;
  const unsigned* wrow0 = Wp + (size_t)nn0 * Kpu + 4 * half;
  const unsigned* wrow1 = Wp + (size_t)nn1 * Kpu + 4 * half;
  int Kp = ((K + 31) >> 5) << 5;             // padded K
  AccFrag a0x, a0y, a1x, a1y;
#pragma unroll
  for (int j = 0; j < 8; ++j) { a0x.f[j] = 0.f; a0y.f[j] = 0.f; a1x.f[j] = 0.f; a1y.f[j] = 0.f; }
  int kt = 0;
  for (; kt + 64 <= Kp; kt += 64) {
    ABfrag aA, aB, w0A, w0B, w1A, w1B;
    *(uint4*)&aA.u[0]  = *(const uint4*)(arow + kt);
    *(uint4*)&aA.u[4]  = *(const uint4*)(arow + kt + 16);
    *(uint4*)&aB.u[0]  = *(const uint4*)(arow + kt + 32);
    *(uint4*)&aB.u[4]  = *(const uint4*)(arow + kt + 48);
    *(uint4*)&w0A.u[0] = *(const uint4*)(wrow0 + (kt >> 1));
    *(uint4*)&w0A.u[4] = *(const uint4*)(wrow0 + (kt >> 1) + 8);
    *(uint4*)&w0B.u[0] = *(const uint4*)(wrow0 + (kt >> 1) + 16);
    *(uint4*)&w0B.u[4] = *(const uint4*)(wrow0 + (kt >> 1) + 24);
    *(uint4*)&w1A.u[0] = *(const uint4*)(wrow1 + (kt >> 1));
    *(uint4*)&w1A.u[4] = *(const uint4*)(wrow1 + (kt >> 1) + 8);
    *(uint4*)&w1B.u[0] = *(const uint4*)(wrow1 + (kt >> 1) + 16);
    *(uint4*)&w1B.u[4] = *(const uint4*)(wrow1 + (kt >> 1) + 24);
    a0x.v = WMMA_BF16(aA.v, w0A.v, a0x.v);
    a1x.v = WMMA_BF16(aA.v, w1A.v, a1x.v);
    a0y.v = WMMA_BF16(aB.v, w0B.v, a0y.v);
    a1y.v = WMMA_BF16(aB.v, w1B.v, a1y.v);
  }
  if (kt < Kp) {                             // single remaining 32-chunk
    ABfrag aA, w0A, w1A;
    *(uint4*)&aA.u[0]  = *(const uint4*)(arow + kt);
    *(uint4*)&aA.u[4]  = *(const uint4*)(arow + kt + 16);
    *(uint4*)&w0A.u[0] = *(const uint4*)(wrow0 + (kt >> 1));
    *(uint4*)&w0A.u[4] = *(const uint4*)(wrow0 + (kt >> 1) + 8);
    *(uint4*)&w1A.u[0] = *(const uint4*)(wrow1 + (kt >> 1));
    *(uint4*)&w1A.u[4] = *(const uint4*)(wrow1 + (kt >> 1) + 8);
    a0x.v = WMMA_BF16(aA.v, w0A.v, a0x.v);
    a1x.v = WMMA_BF16(aA.v, w1A.v, a1x.v);
  }
  AccFrag acc0, acc1;
  acc0.v = a0x.v + a0y.v;
  acc1.v = a1x.v + a1y.v;
#pragma unroll
  for (int t2 = 0; t2 < 2; ++t2) {
    int n = t2 ? n1 : n0;
    const AccFrag& ac = t2 ? acc1 : acc0;
    if (n < N) {
      float bn = bias ? bias[n] : 0.f;
#pragma unroll
      for (int j = 0; j < 8; ++j) {
        int mm = tm * 16 + j + 8 * half;
        float v = ac.f[j] + bn;
        if (res) v += res[(size_t)mm * ldo + n];
        if (act == 1) v = fmaxf(v, 0.f);
        else if (act == 2) v = 0.5f * v * (1.f + erff(v * 0.70710678118654752f));
        if (outf) outf[(size_t)mm * ldo + n] = v;
        if (outb) outb[(size_t)mm * ldob + n] = (unsigned short)rne_bf16_bits(v);
      }
    }
  }
}

// ---------------- LayerNorm over 192 features -> bf16 mirror ----------------
__global__ __launch_bounds__(256)
void ln_kernel(const float* __restrict__ x, const float* __restrict__ g,
               const float* __restrict__ b, unsigned short* __restrict__ outb)
{
  __shared__ float buf[256];
  int row = blockIdx.x, tid = threadIdx.x;
  float v = (tid < HID) ? x[(size_t)row * HID + tid] : 0.f;
  buf[tid] = v; __syncthreads();
  for (int s = 128; s > 0; s >>= 1) { if (tid < s) buf[tid] += buf[tid + s]; __syncthreads(); }
  float mean = buf[0] / (float)HID; __syncthreads();
  float d = (tid < HID) ? (v - mean) : 0.f;
  buf[tid] = d * d; __syncthreads();
  for (int s = 128; s > 0; s >>= 1) { if (tid < s) buf[tid] += buf[tid + s]; __syncthreads(); }
  float var = buf[0] / (float)HID;
  float rs = rsqrtf(var + 1e-6f);
  if (tid < HID)
    outb[(size_t)row * HID + tid] =
        (unsigned short)rne_bf16_bits((v - mean) * rs * g[tid] + b[tid]);
}

// ---------------- tiny attention: one block per (batch, head), T = 40 ----------------
__global__ __launch_bounds__(64)
void attn_kernel(const float* __restrict__ qkv, unsigned short* __restrict__ ob)
{
  int bh = blockIdx.x;
  int b = bh / HEADS, h = bh - b * HEADS;
  int t = threadIdx.x;
  if (t >= NC) return;
  const float* base = qkv + (size_t)(b * NC) * QKVD;
  float q[HD];
#pragma unroll
  for (int d = 0; d < HD; ++d) q[d] = base[t * QKVD + h * HD + d];
  float s[NC]; float mx = -1e30f;
  for (int tk = 0; tk < NC; ++tk) {
    float dot = 0.f;
#pragma unroll
    for (int d = 0; d < HD; ++d) dot += q[d] * base[tk * QKVD + HID + h * HD + d];
    dot *= 0.17677669529663688f;   // 32^-0.5
    s[tk] = dot; mx = fmaxf(mx, dot);
  }
  float sum = 0.f;
  for (int tk = 0; tk < NC; ++tk) { s[tk] = expf(s[tk] - mx); sum += s[tk]; }
  float inv = 1.f / sum;
  float acc[HD];
#pragma unroll
  for (int d = 0; d < HD; ++d) acc[d] = 0.f;
  for (int tk = 0; tk < NC; ++tk) {
    float wv = s[tk] * inv;
#pragma unroll
    for (int d = 0; d < HD; ++d) acc[d] += wv * base[tk * QKVD + 2 * HID + h * HD + d];
  }
#pragma unroll
  for (int d = 0; d < HD; ++d)
    ob[(size_t)(b * NC + t) * HID + h * HD + d] = (unsigned short)rne_bf16_bits(acc[d]);
}

// ---------------- sin embed: [R x 3] f32 -> [R x 64] bf16 (51 used, padded) ----------------
__global__ void sinembed_kernel(const float* __restrict__ src, unsigned short* __restrict__ outb,
                                const float* __restrict__ basis, int R)
{
  int r = blockIdx.x * blockDim.x + threadIdx.x;
  if (r >= R) return;
  float x0 = src[r * 3], x1 = src[r * 3 + 1], x2 = src[r * 3 + 2];
  unsigned short* o = outb + (size_t)r * 64;
  o[0] = (unsigned short)rne_bf16_bits(x0);
  o[1] = (unsigned short)rne_bf16_bits(x1);
  o[2] = (unsigned short)rne_bf16_bits(x2);
#pragma unroll
  for (int e = 0; e < 24; ++e) {
    float p = x0 * basis[e] + x1 * basis[24 + e] + x2 * basis[48 + e];
    o[3 + e]  = (unsigned short)rne_bf16_bits(sinf(p));
    o[27 + e] = (unsigned short)rne_bf16_bits(cosf(p));
  }
#pragma unroll
  for (int k = MSGD; k < 64; ++k) o[k] = 0;
}

__global__ void add_kernel(float* __restrict__ dst, const float* __restrict__ a,
                           const float* __restrict__ b, int n)
{
  int i = blockIdx.x * blockDim.x + threadIdx.x;
  if (i < n) dst[i] = a[i] + b[i];
}

__global__ void fill_u32_kernel(unsigned* dst, unsigned v, int n)
{
  int i = blockIdx.x * blockDim.x + threadIdx.x;
  if (i < n) dst[i] = v;
}

__global__ void concat_kernel(const float* __restrict__ centers, const float* __restrict__ dxyz,
                              const float* __restrict__ dtxyz, const float* __restrict__ sh2d,
                              const float* __restrict__ sh3d, float* __restrict__ out)
{
  int r = blockIdx.x * blockDim.x + threadIdx.x;
  if (r >= BCC) return;
  float* o = out + (size_t)r * 62;
#pragma unroll
  for (int d = 0; d < 3; ++d) {
    o[d]     = centers[r * 3 + d] + dxyz[r * 3 + d];
    o[3 + d] = centers[r * 3 + d] + dtxyz[r * 3 + d];
  }
#pragma unroll
  for (int j = 0; j < 7; ++j) o[6 + j] = sh2d[r * 7 + j];
#pragma unroll
  for (int j = 0; j < 49; ++j) o[13 + j] = sh3d[r * 49 + j];
}

// ---------------- host orchestration ----------------
struct PW { const unsigned* p; int kpu; };

extern "C" void kernel_launch(void* const* d_in, const int* in_sizes, int n_in,
                              void* d_out, int out_size, void* d_ws, size_t ws_size,
                              hipStream_t stream)
{
  (void)in_sizes; (void)n_in; (void)out_size; (void)ws_size;
  const float* pc  = (const float*)d_in[0];
  const int*   idx = (const int*)d_in[1];
  auto F = [&](int i) { return (const float*)d_in[i]; };
  // params flattened depth-first in dict insertion order:
  const int I_EMBED = 2;    // embed: w[51,192], b[192]
  const int I_LOCAL = 4;    // 4 lins (w,b): [51,96],[96,96],[96,192],[192,192]
  const int I_GLOB  = 12;   // 3 lins (w,b): [192,192] each
  const int I_VIT   = 18;   // 4 vits * 6 blocks * 12 leaves
  const int I_LNP   = 306;  // 4 * (g,b)
  const int I_HEAD  = 314;  // 4 heads * 4 lins * 2
  const int I_SH2DE = 346;  // sh2d_embed: w[7,192], b[192]
  const int I_BASIS = 348;  // [3,24]
  const float* basis = F(I_BASIS);

  // workspace carve
  char* wp_ = (char*)d_ws;
  auto carve = [&](size_t bytes) -> char* {
    char* r = wp_; wp_ += (bytes + 255) & ~(size_t)255; return r;
  };
  int*            nbr     = (int*)           carve((size_t)BCC * KNN * 4);
  float*          centers = (float*)         carve((size_t)BCC * 3 * 4);
  unsigned*       featk   = (unsigned*)      carve((size_t)BCC * HID * 4);
  unsigned short* feat0b  = (unsigned short*)carve((size_t)BCC * HID * 2);
  float*          featf   = (float*)         carve((size_t)BCC * HID * 4);
  float*          pos     = (float*)         carve((size_t)BCC * HID * 4);
  float*          x       = (float*)         carve((size_t)BCC * HID * 4);
  unsigned short* lnbb    = (unsigned short*)carve((size_t)BCC * HID * 2);
  float*          qkvb    = (float*)         carve((size_t)BCC * QKVD * 4);
  unsigned short* attob   = (unsigned short*)carve((size_t)BCC * HID * 2);
  unsigned short* mlpbb   = (unsigned short*)carve((size_t)BCC * FFD * 2);
  unsigned short* emb51b  = (unsigned short*)carve((size_t)BCC * 64 * 2);
  unsigned short* tmpAb   = (unsigned short*)carve((size_t)BCC * HID * 2);
  unsigned short* tmpBb   = (unsigned short*)carve((size_t)BCC * HID * 2);
  float*          tmpF    = (float*)         carve((size_t)BCC * HID * 4);
  float*          dxyz    = (float*)         carve((size_t)BCC * 3 * 4);
  float*          dtxyz   = (float*)         carve((size_t)BCC * 3 * 4);
  float*          sh2df   = (float*)         carve((size_t)BCC * 7 * 4);
  unsigned short* sh2db   = (unsigned short*)carve((size_t)BCC * 32 * 2);
  float*          sh3d    = (float*)         carve((size_t)BCC * 49 * 4);

  // ---- pre-pack all weights into fragment-native bf16-pair layout ----
  auto packw = [&](int fi, int K, int N) -> PW {
    int kpu = ((K + 31) / 32) * 16;
    int total = N * kpu;
    unsigned* p = (unsigned*)carve((size_t)total * 4);
    pack_weight_kernel<<<(total + 255) / 256, 256, 0, stream>>>(F(fi), p, K, N, kpu, total);
    return PW{p, kpu};
  };
  PW pw_embed = packw(I_EMBED, MSGD, HID);
  PW pw_loc[4] = { packw(I_LOCAL + 0, MSGD, 96), packw(I_LOCAL + 2, 96, 96),
                   packw(I_LOCAL + 4, 96, HID),  packw(I_LOCAL + 6, HID, HID) };
  PW pw_glob[3] = { packw(I_GLOB + 0, HID, HID), packw(I_GLOB + 2, HID, HID),
                    packw(I_GLOB + 4, HID, HID) };
  PW pw_qkv[24], pw_proj[24], pw_fc1[24], pw_fc2[24];
  for (int i = 0; i < 24; ++i) {
    int pb = I_VIT + i * 12;
    pw_qkv[i]  = packw(pb + 2, HID, QKVD);
    pw_proj[i] = packw(pb + 4, HID, HID);
    pw_fc1[i]  = packw(pb + 8, HID, FFD);
    pw_fc2[i]  = packw(pb + 10, FFD, HID);
  }
  static const int headDout[4] = {3, 3, 7, 49};
  PW pw_head[4][4];
  for (int h = 0; h < 4; ++h) {
    int hb = I_HEAD + h * 8;
    pw_head[h][0] = packw(hb + 0, HID, HID);
    pw_head[h][1] = packw(hb + 2, HID, 96);
    pw_head[h][2] = packw(hb + 4, 96, 96);
    pw_head[h][3] = packw(hb + 6, 96, headDout[h]);
  }
  PW pw_sh2de = packw(I_SH2DE, 7, HID);

  auto gemm = [&](const unsigned short* Ab, int ldab, const PW& w, const float* bias,
                  const float* res, float* outf, int ldo, unsigned short* outb, int ldob,
                  int M, int K, int N, int act) {
    int Mt = (M + 15) / 16, Nt2 = (N + 31) / 32;
    int tiles = Mt * Nt2;
    int blocks = (tiles + 7) / 8;
    gemm_wmma_kernel<<<blocks, 256, 0, stream>>>(Ab, ldab, w.p, w.kpu, bias, res,
                                                 outf, ldo, outb, ldob, M, K, N, act, Nt2, tiles);
  };

  // 1. KNN + center gather
  knn_kernel<<<BCC, 256, NPTS * sizeof(unsigned), stream>>>(pc, idx, nbr, centers);

  // 2. fused PointConv local MLP with max aggregation (featk holds order-keys)
  fill_u32_kernel<<<(BCC * HID + 255) / 256, 256, 0, stream>>>(featk, 0u, BCC * HID);
  fill_u32_kernel<<<(BCC * 32 / 2 + 255) / 256, 256, 0, stream>>>((unsigned*)sh2db, 0u,
                                                                  BCC * 32 / 2);
  {
    int tiles = BCC * 26;
    pointconv_kernel<<<(tiles + 3) / 4, 128, 0, stream>>>(
        pc, nbr, centers, basis,
        pw_loc[0].p, F(I_LOCAL + 1), pw_loc[1].p, F(I_LOCAL + 3),
        pw_loc[2].p, F(I_LOCAL + 5), pw_loc[3].p, F(I_LOCAL + 7),
        featk, tiles);
  }
  decode_feat_kernel<<<(BCC * HID + 255) / 256, 256, 0, stream>>>(featk, feat0b, BCC * HID);

  // 3. global MLP: lin,relu,lin,relu,lin  (bf16 chain, f32 only at the end)
  gemm(feat0b, HID, pw_glob[0], F(I_GLOB + 1), nullptr, nullptr, 0, tmpAb, HID, BCC, HID, HID, 1);
  gemm(tmpAb, HID, pw_glob[1], F(I_GLOB + 3), nullptr, nullptr, 0, tmpBb, HID, BCC, HID, HID, 1);
  gemm(tmpBb, HID, pw_glob[2], F(I_GLOB + 5), nullptr, featf, HID, nullptr, 0, BCC, HID, HID, 0);

  // 4. pos = embed(concat(centers, sin_embed(centers)))
  sinembed_kernel<<<(BCC + 255) / 256, 256, 0, stream>>>(centers, emb51b, basis, BCC);
  gemm(emb51b, 64, pw_embed, F(I_EMBED + 1), nullptr, pos, HID, nullptr, 0, BCC, MSGD, HID, 0);

  auto run_vit = [&](int v) {
    add_kernel<<<(BCC * HID + 255) / 256, 256, 0, stream>>>(x, featf, pos, BCC * HID);
    for (int bl = 0; bl < DEPTH; ++bl) {
      int i = v * DEPTH + bl;
      int pb = I_VIT + i * 12;
      ln_kernel<<<BCC, 256, 0, stream>>>(x, F(pb + 0), F(pb + 1), lnbb);
      gemm(lnbb, HID, pw_qkv[i], F(pb + 3), nullptr, qkvb, QKVD, nullptr, 0, BCC, HID, QKVD, 0);
      attn_kernel<<<BATCH * HEADS, 64, 0, stream>>>(qkvb, attob);
      gemm(attob, HID, pw_proj[i], F(pb + 5), x, x, HID, nullptr, 0, BCC, HID, HID, 0);
      ln_kernel<<<BCC, 256, 0, stream>>>(x, F(pb + 6), F(pb + 7), lnbb);
      gemm(lnbb, HID, pw_fc1[i], F(pb + 9), nullptr, nullptr, 0, mlpbb, FFD, BCC, HID, FFD, 2);
      gemm(mlpbb, FFD, pw_fc2[i], F(pb + 11), x, x, HID, nullptr, 0, BCC, FFD, HID, 0);
    }
  };
  auto run_head = [&](int h, float* outv, unsigned short* outvb, int ldob, int dout) {
    int hb = I_HEAD + h * 8;
    gemm(lnbb, HID, pw_head[h][0], F(hb + 1), nullptr, nullptr, 0, tmpAb, HID, BCC, HID, HID, 1);
    gemm(tmpAb, HID, pw_head[h][1], F(hb + 3), nullptr, nullptr, 0, tmpBb, 96, BCC, HID, 96, 1);
    gemm(tmpBb, 96, pw_head[h][2], F(hb + 5), nullptr, nullptr, 0, tmpAb, 96, BCC, 96, 96, 1);
    gemm(tmpAb, 96, pw_head[h][3], F(hb + 7), nullptr, outv, dout, outvb, ldob, BCC, 96, dout, 0);
  };

  // 5. four cascaded ViT decoders with accumulating positional embeddings
  run_vit(0);
  ln_kernel<<<BCC, 256, 0, stream>>>(x, F(I_LNP + 0), F(I_LNP + 1), lnbb);
  run_head(0, dxyz, nullptr, 0, 3);
  sinembed_kernel<<<(BCC + 255) / 256, 256, 0, stream>>>(dxyz, emb51b, basis, BCC);
  gemm(emb51b, 64, pw_embed, F(I_EMBED + 1), nullptr, tmpF, HID, nullptr, 0, BCC, MSGD, HID, 0);
  add_kernel<<<(BCC * HID + 255) / 256, 256, 0, stream>>>(pos, pos, tmpF, BCC * HID);

  run_vit(1);
  ln_kernel<<<BCC, 256, 0, stream>>>(x, F(I_LNP + 2), F(I_LNP + 3), lnbb);
  run_head(1, dtxyz, nullptr, 0, 3);
  sinembed_kernel<<<(BCC + 255) / 256, 256, 0, stream>>>(dtxyz, emb51b, basis, BCC);
  gemm(emb51b, 64, pw_embed, F(I_EMBED + 1), nullptr, tmpF, HID, nullptr, 0, BCC, MSGD, HID, 0);
  add_kernel<<<(BCC * HID + 255) / 256, 256, 0, stream>>>(pos, pos, tmpF, BCC * HID);

  run_vit(2);
  ln_kernel<<<BCC, 256, 0, stream>>>(x, F(I_LNP + 4), F(I_LNP + 5), lnbb);
  run_head(2, sh2df, sh2db, 32, 7);
  gemm(sh2db, 32, pw_sh2de, F(I_SH2DE + 1), nullptr, tmpF, HID, nullptr, 0, BCC, 7, HID, 0);
  add_kernel<<<(BCC * HID + 255) / 256, 256, 0, stream>>>(pos, pos, tmpF, BCC * HID);

  run_vit(3);
  ln_kernel<<<BCC, 256, 0, stream>>>(x, F(I_LNP + 6), F(I_LNP + 7), lnbb);
  run_head(3, sh3d, nullptr, 0, 49);

  // 6. final concat -> [B, C, 62]
  concat_kernel<<<(BCC + 255) / 256, 256, 0, stream>>>(centers, dxyz, dtxyz, sh2df, sh3d,
                                                       (float*)d_out);
}
